// Metric_Learning_based_Adjacency_Matrix_5334349382038
// MI455X (gfx1250) — matrix operand order repaired and checked by reading, not compiled, not run on hardware
//
#include <hip/hip_runtime.h>
#include <hip/hip_bf16.h>
#include <stdint.h>

#define BSZ 4096
#define DD  2048
#define RR  2048

// LDS panel geometry: 256 rows of 32 bf16 (64B payload) at 80B pitch
#define PITCH  80
#define PANEL  (256 * PITCH)      // 20480 B
#define BUFSZ  (2 * PANEL)        // A + B panel, one buffer
#define LDS_TOTAL (2 * BUFSZ)     // double buffered = 81920 B

typedef __attribute__((ext_vector_type(16))) __bf16 bf16x16;
typedef __attribute__((ext_vector_type(8)))  float  f32x8;

union FragU { bf16x16 v; uint4 u[2]; };

__device__ __forceinline__ unsigned short f32_to_bf16(float x) {
    unsigned u = __float_as_uint(x);
    u = u + 0x7FFFu + ((u >> 16) & 1u);   // round-to-nearest-even
    return (unsigned short)(u >> 16);
}
__device__ __forceinline__ float bf16_to_f32(unsigned short h) {
    return __uint_as_float(((unsigned)h) << 16);
}

__device__ __forceinline__ f32x8 zero8() {
    f32x8 z;
#pragma unroll
    for (int e = 0; e < 8; e++) z[e] = 0.0f;
    return z;
}

// ---- CDNA5 async global->LDS copy of one 256x32 bf16 panel (512 threads) ----
// Each thread issues 2x GLOBAL_LOAD_ASYNC_TO_LDS_B128 (ASYNCcnt-tracked).
__device__ __forceinline__ void async_copy_panel(unsigned char* dst,
                                                 const unsigned short* __restrict__ src,
                                                 int ld, int row0, int k) {
    int tid = threadIdx.x;
#pragma unroll
    for (int i = 0; i < 2; i++) {
        int idx = tid + i * 512;          // 0..1023
        int row = idx >> 2;               // 0..255
        int q   = idx & 3;                // 16B quarter of the 64B row payload
        unsigned int lds_off = (unsigned int)(uintptr_t)(dst + row * PITCH + q * 16);
        const unsigned short* g = src + (size_t)(row0 + row) * ld + k + q * 8;
        asm volatile("global_load_async_to_lds_b128 %0, %1, off"
                     :: "v"(lds_off), "v"((unsigned long long)(uintptr_t)g)
                     : "memory");
    }
}
__device__ __forceinline__ void wait_async_zero() {
    asm volatile("s_wait_asynccnt 0x0" ::: "memory");
}

// A fragment (16x32 bf16, MxK) from LDS panel: lane m = lane&15, hi = lane>>4
// e=0..7 -> K=8*hi+e (bytes 16*hi..+16) ; e=8..15 -> K=16+8*hi+e-8 (bytes 32+16*hi..+16)
__device__ __forceinline__ bf16x16 lds_frag_a(const unsigned char* sA, int row0) {
    int lane = threadIdx.x & 31;
    int m = lane & 15, hi = lane >> 4;
    const unsigned char* p = sA + (row0 + m) * PITCH + 16 * hi;
    FragU f;
    f.u[0] = *(const uint4*)(p);
    f.u[1] = *(const uint4*)(p + 32);
    return f.v;
}
// B fragment (32x16 bf16, KxN) from transposed panel: lane n, e=0..7 -> K=8*hi+e ; e=8..15 -> K=16+8*hi+e-8
__device__ __forceinline__ bf16x16 lds_frag_b(const unsigned char* sB, int row0) {
    int lane = threadIdx.x & 31;
    int n = lane & 15, hi = lane >> 4;
    const unsigned char* p = sB + (row0 + n) * PITCH + 16 * hi;
    FragU f;
    f.u[0] = *(const uint4*)(p);
    f.u[1] = *(const uint4*)(p + 32);
    return f.v;
}

// ---- 256x256 block tile, 16 waves (4x4), wave tile 64x64, async double-buffer ----
__device__ __forceinline__ void gemm_core_256(const unsigned short* __restrict__ Abase, int lda,
                                              const unsigned short* __restrict__ Btbase, int ldb,
                                              int blockM0, int blockN0, int K,
                                              unsigned char* lds,
                                              f32x8 (&acc)[4][4], int& wmo, int& wno) {
    int wid = threadIdx.x >> 5;
    int wml = (wid >> 2) * 64;            // wave M offset inside block
    int wnl = (wid & 3) * 64;             // wave N offset inside block
    wmo = blockM0 + wml;
    wno = blockN0 + wnl;

#pragma unroll
    for (int i = 0; i < 4; i++)
#pragma unroll
        for (int j = 0; j < 4; j++) acc[i][j] = zero8();

    unsigned char* buf0 = lds;
    unsigned char* buf1 = lds + BUFSZ;

    // prologue: fill buffer 0
    async_copy_panel(buf0,         Abase,  lda, blockM0, 0);
    async_copy_panel(buf0 + PANEL, Btbase, ldb, blockN0, 0);
    wait_async_zero();
    __syncthreads();

    int cur = 0;
    for (int k = 0; k < K; k += 32) {
        unsigned char* curb = cur ? buf1 : buf0;
        unsigned char* nxtb = cur ? buf0 : buf1;
        if (k + 32 < K) {
            async_copy_panel(nxtb,         Abase,  lda, blockM0, k + 32);
            async_copy_panel(nxtb + PANEL, Btbase, ldb, blockN0, k + 32);
        }
        bf16x16 a[4], b[4];
#pragma unroll
        for (int t = 0; t < 4; t++) a[t] = lds_frag_a(curb,         wml + 16 * t);
#pragma unroll
        for (int t = 0; t < 4; t++) b[t] = lds_frag_b(curb + PANEL, wnl + 16 * t);
#pragma unroll
        for (int i = 0; i < 4; i++)
#pragma unroll
            for (int j = 0; j < 4; j++)
                acc[i][j] = __builtin_amdgcn_wmma_f32_16x16x32_bf16(
                    false, a[i], false, b[j], (short)0, acc[i][j], false, false);
        wait_async_zero();
        __syncthreads();
        cur ^= 1;
    }
}

// ---------------- convert / transpose ----------------

__global__ void convert_f_kernel(const float* __restrict__ in,
                                 unsigned short* __restrict__ out, int n4) {
    int idx = blockIdx.x * blockDim.x + threadIdx.x;
    if (idx < n4) {
        float4 v = ((const float4*)in)[idx];
        ushort4 o;
        o.x = f32_to_bf16(v.x); o.y = f32_to_bf16(v.y);
        o.z = f32_to_bf16(v.z); o.w = f32_to_bf16(v.w);
        ((ushort4*)out)[idx] = o;
    }
}

__global__ __launch_bounds__(256) void transpose_p_kernel(const float* __restrict__ P,
                                                          unsigned short* __restrict__ Ptb) {
    __shared__ unsigned short tile[32][33];
    int tx = threadIdx.x, ty = threadIdx.y;   // (32, 8)
    int bd = blockIdx.y * 32, br = blockIdx.x * 32;
#pragma unroll
    for (int i = 0; i < 4; i++) {
        int d = bd + ty + 8 * i;
        tile[ty + 8 * i][tx] = f32_to_bf16(P[(size_t)d * RR + br + tx]);
    }
    __syncthreads();
#pragma unroll
    for (int i = 0; i < 4; i++) {
        int r = br + ty + 8 * i;
        Ptb[(size_t)r * DD + bd + tx] = tile[tx][ty + 8 * i];
    }
}

// ---------------- GEMM 1: Z = F @ P ----------------

__global__ __launch_bounds__(512) void gemm_z_kernel(const unsigned short* __restrict__ Fb,
                                                     const unsigned short* __restrict__ Ptb,
                                                     unsigned short* __restrict__ Zb) {
    extern __shared__ unsigned char lds[];
    f32x8 acc[4][4];
    int m0, n0;
    gemm_core_256(Fb, DD, Ptb, DD, blockIdx.y * 256, blockIdx.x * 256, DD, lds, acc, m0, n0);

    int lane = threadIdx.x & 31;
    int n = lane & 15, hi = lane >> 4;
#pragma unroll
    for (int i = 0; i < 4; i++)
#pragma unroll
        for (int j = 0; j < 4; j++)
#pragma unroll
            for (int r = 0; r < 8; r++) {
                int row = m0 + 16 * i + r + 8 * hi;
                int col = n0 + 16 * j + n;
                Zb[(size_t)row * RR + col] = f32_to_bf16(acc[i][j][r]);
            }
}

// ---------------- sq_norm (deterministic tree reduce) ----------------

__global__ __launch_bounds__(128) void sqnorm_kernel(const unsigned short* __restrict__ Zb,
                                                     float* __restrict__ sqn) {
    __shared__ float red[128];
    int row = blockIdx.x;
    float s = 0.0f;
    for (int c = threadIdx.x; c < RR; c += 128) {
        float z = bf16_to_f32(Zb[(size_t)row * RR + c]);
        s += z * z;
    }
    red[threadIdx.x] = s;
    __syncthreads();
    for (int off = 64; off > 0; off >>= 1) {
        if ((int)threadIdx.x < off) red[threadIdx.x] += red[threadIdx.x + off];
        __syncthreads();
    }
    if (threadIdx.x == 0) sqn[row] = red[0];
}

// ---------------- GEMM 2: G = Z @ Z^T -> adjacency_pre ----------------

__global__ __launch_bounds__(512) void gemm_adj_kernel(const unsigned short* __restrict__ Zb,
                                                       const float* __restrict__ sqn,
                                                       float* __restrict__ adj) {
    extern __shared__ unsigned char lds[];
    f32x8 acc[4][4];
    int m0, n0;
    gemm_core_256(Zb, RR, Zb, RR, blockIdx.y * 256, blockIdx.x * 256, RR, lds, acc, m0, n0);

    int lane = threadIdx.x & 31;
    int n = lane & 15, hi = lane >> 4;
#pragma unroll
    for (int i = 0; i < 4; i++)
#pragma unroll
        for (int j = 0; j < 4; j++)
#pragma unroll
            for (int r = 0; r < 8; r++) {
                int row = m0 + 16 * i + r + 8 * hi;
                int col = n0 + 16 * j + n;
                float g = acc[i][j][r];
                float s = fmaxf(sqn[row] + sqn[col] - 2.0f * g, 0.0f);
                float dist = (row == col) ? 0.0f : sqrtf(s);
                adj[(size_t)row * BSZ + col] = __expf(-dist);   // ALPHA = 1
            }
}

// ---------------- degree row-sums + normalization ----------------

__global__ __launch_bounds__(256) void rowsum_kernel(const float* __restrict__ adj,
                                                     float* __restrict__ rs) {
    __shared__ float red[256];
    int row = blockIdx.x;
    float s = 0.0f;
    for (int c = threadIdx.x; c < BSZ; c += 256)
        s += adj[(size_t)row * BSZ + c];
    red[threadIdx.x] = s;
    __syncthreads();
    for (int off = 128; off > 0; off >>= 1) {
        if ((int)threadIdx.x < off) red[threadIdx.x] += red[threadIdx.x + off];
        __syncthreads();
    }
    if (threadIdx.x == 0) rs[row] = red[0];
}

__global__ __launch_bounds__(256) void normalize_kernel(const float* __restrict__ adj,
                                                        const float* __restrict__ rs,
                                                        float* __restrict__ outN) {
    size_t idx = (size_t)blockIdx.x * blockDim.x + threadIdx.x;   // float4 index
    size_t e0 = idx * 4;
    size_t i = e0 / BSZ, j0 = e0 % BSZ;
    float4 a = ((const float4*)adj)[idx];
    float ri = rsqrtf(rs[i]);
    float4 o;
    o.x = ri * a.x * rsqrtf(rs[j0 + 0]);
    o.y = ri * a.y * rsqrtf(rs[j0 + 1]);
    o.z = ri * a.z * rsqrtf(rs[j0 + 2]);
    o.w = ri * a.w * rsqrtf(rs[j0 + 3]);
    ((float4*)outN)[idx] = o;
}

// ---------------- launcher ----------------

extern "C" void kernel_launch(void* const* d_in, const int* in_sizes, int n_in,
                              void* d_out, int out_size, void* d_ws, size_t ws_size,
                              hipStream_t stream) {
    (void)in_sizes; (void)n_in; (void)out_size; (void)ws_size;
    const float* F = (const float*)d_in[0];   // [4096, 2048]
    const float* P = (const float*)d_in[1];   // [2048, 2048]
    float* outN = (float*)d_out;                        // normalized  [4096,4096]
    float* adj  = (float*)d_out + (size_t)BSZ * BSZ;    // adjacency_pre

    unsigned char* ws = (unsigned char*)d_ws;
    unsigned short* Fb  = (unsigned short*)(ws);                              // 16 MB
    unsigned short* Ptb = (unsigned short*)(ws + (size_t)16 * 1024 * 1024);   //  8 MB
    unsigned short* Zb  = (unsigned short*)(ws + (size_t)24 * 1024 * 1024);   // 16 MB
    float* sqn = (float*)(ws + (size_t)40 * 1024 * 1024);                     // 16 KB
    float* rs  = sqn + BSZ;                                                   // 16 KB

    convert_f_kernel<<<(BSZ * DD / 4 + 255) / 256, 256, 0, stream>>>(F, Fb, BSZ * DD / 4);
    transpose_p_kernel<<<dim3(RR / 32, DD / 32), dim3(32, 8), 0, stream>>>(P, Ptb);
    gemm_z_kernel<<<dim3(RR / 256, BSZ / 256), 512, LDS_TOTAL, stream>>>(Fb, Ptb, Zb);
    sqnorm_kernel<<<BSZ, 128, 0, stream>>>(Zb, sqn);
    gemm_adj_kernel<<<dim3(BSZ / 256, BSZ / 256), 512, LDS_TOTAL, stream>>>(Zb, sqn, adj);
    rowsum_kernel<<<BSZ, 256, 0, stream>>>(adj, rs);
    normalize_kernel<<<(unsigned)((size_t)BSZ * BSZ / 4 / 256), 256, 0, stream>>>(adj, rs, outN);
}
